// PerformerAttention_84524956385938
// MI455X (gfx1250) — matrix operand order repaired and checked by reading, not compiled, not run on hardware
//
#include <hip/hip_runtime.h>
#include <hip/hip_bf16.h>
#include <math.h>

// ---------------------------------------------------------------------------
// Performer (linear) attention forward for MI455X / gfx1250 (wave32, WMMA).
// All contractions use v_wmma_f32_16x16x32_bf16 with hi/lo bf16 split
// (hi*hi + hi*lo + lo*hi) for ~fp32 accuracy, f32 accumulation.
// ---------------------------------------------------------------------------

typedef __attribute__((ext_vector_type(16))) __bf16       v16bf;
typedef __attribute__((ext_vector_type(8)))  float        v8f;
typedef __attribute__((ext_vector_type(4)))  unsigned int u32x4;
typedef __attribute__((ext_vector_type(4)))  float        f32x4;

constexpr int NTOK = 2048, DIMM = 1024, NH = 16, HD = 64, NF = 64;
constexpr int CHK = 32, NCHK = NTOK / CHK; // 64 chunks

constexpr size_t E_X  = (size_t)NTOK * DIMM;          // 2M elems
constexpr size_t E_W  = (size_t)DIMM * DIMM;          // 1M
constexpr size_t E_OM = (size_t)NF * HD;              // 4096
constexpr size_t E_H  = (size_t)NH * NTOK * HD;       // 2M
constexpr size_t E_KV = (size_t)NH * NCHK * NF * HD;  // 4M
constexpr size_t E_KS = (size_t)NH * NCHK * NF;       // 64K
constexpr size_t E_AT = (size_t)NTOK * DIMM;          // 2M

constexpr size_t MB = 1024ull * 1024ull;
// Fresh allocations (72.02 MB total)
constexpr size_t O_XHI  = 0 * MB,  O_XLO  = 4 * MB;
constexpr size_t O_WQHI = 8 * MB,  O_WQLO = 10 * MB;
constexpr size_t O_WKHI = 12 * MB, O_WKLO = 14 * MB;
constexpr size_t O_WVHI = 16 * MB, O_WVLO = 18 * MB;
constexpr size_t O_WOHI = 20 * MB, O_WOLO = 22 * MB;
constexpr size_t O_OMHI = 24 * MB, O_OMLO = 24 * MB + 8192;
constexpr size_t O_QHI  = 25 * MB, O_QLO  = 29 * MB;
constexpr size_t O_KHI  = 33 * MB, O_KLO  = 37 * MB;
constexpr size_t O_VTHI = 41 * MB, O_VTLO = 45 * MB;
constexpr size_t O_PQHI = 49 * MB, O_PQLO = 53 * MB;
constexpr size_t O_PKHI = 57 * MB, O_PKLO = 61 * MB;
constexpr size_t O_PKTHI = 65 * MB, O_PKTLO = 69 * MB;
constexpr size_t WS_NEEDED = 73 * MB;
// Aliases onto dead regions (write-after-last-read ordering is guaranteed
// by the launch sequence below):
constexpr size_t O_CKV   = O_XHI;   // 16MB f32 chunk-KV  (X, Wq, Wk dead after QKV)
constexpr size_t O_CKS   = O_WVHI;  // 256KB f32 chunk ksum (Wv hi dead)
constexpr size_t O_KSP   = O_WVLO;  // 256KB f32 ksum prefix (Wv lo dead)
constexpr size_t O_KVTHI = O_QHI;   // 8MB KV^T prefix hi (q planes dead after phi)
constexpr size_t O_KVTLO = O_KHI;   // 8MB KV^T prefix lo (k planes dead after phi)
constexpr size_t O_ATHI  = O_PKTHI; // 4MB attn hi (phi_k^T dead after pass1)
constexpr size_t O_ATLO  = O_PKTLO; // 4MB attn lo

#define WMMA_BF16(a, b, c) \
  __builtin_amdgcn_wmma_f32_16x16x32_bf16(false, (a), false, (b), (short)0, (c), false, false)

// A fragment: 16x32 (MxK) bf16, row-major source with leading dim `ld`.
// Lane l holds row (l&15); lanes<16 take K {k0..k0+7, k0+16..k0+23},
// lanes>=16 take K {k0+8..k0+15, k0+24..k0+31} (ISA 16-bit A layout).
__device__ inline v16bf fragA(const __bf16* base, int row, int k0, int ld, int lane) {
  const __bf16* p = base + (size_t)(row + (lane & 15)) * ld + (k0 + ((lane & 16) >> 1));
  union { u32x4 u[2]; v16bf v; } x;
  x.u[0] = *(const u32x4*)p;
  x.u[1] = *(const u32x4*)(p + 16);
  return x.v;
}

// B fragment: 32x16 (KxN). Lane l holds column (l&15) = row (col+(l&15)) of the
// row-major transposed operand; lanes<16 take K {k0..k0+7, k0+16..k0+23},
// lanes>=16 take K {k0+8..k0+15, k0+24..k0+31} (matches fragA).
__device__ inline v16bf fragB(const __bf16* base, int col, int k0, int ld, int lane) {
  const __bf16* p = base + (size_t)(col + (lane & 15)) * ld + (k0 + ((lane & 16) >> 1));
  union { u32x4 u[2]; v16bf v; } x;
  x.u[0] = *(const u32x4*)p;
  x.u[1] = *(const u32x4*)(p + 16);
  return x.v;
}

__device__ inline void splitf(float f, __bf16& h, __bf16& l) {
  h = (__bf16)f;
  l = (__bf16)(f - (float)h);
}

// ---------------------------------------------------------------------------
// 0) f32 -> bf16 hi/lo split
// ---------------------------------------------------------------------------
__global__ void k_convert(const float* __restrict__ s, __bf16* __restrict__ hi,
                          __bf16* __restrict__ lo, int n) {
  int i = blockIdx.x * blockDim.x + threadIdx.x;
  if (i < n) {
    float f = s[i];
    __bf16 h = (__bf16)f;
    hi[i] = h;
    lo[i] = (__bf16)(f - (float)h);
  }
}

// ---------------------------------------------------------------------------
// 1) QKV: C[n][o] = sum_k X[n][k] * W[o][k]; block = 8 waves, 128x128 tile,
//    wave tile = 32x64. z selects q/k/v. q,k stored head-major [h][n][d];
//    v stored transposed [h][d][n] (only layout ever consumed).
// ---------------------------------------------------------------------------
__global__ void __launch_bounds__(256) k_gemm_qkv(char* ws) {
  const __bf16* Xhi = (const __bf16*)(ws + O_XHI);
  const __bf16* Xlo = (const __bf16*)(ws + O_XLO);
  int z = blockIdx.z;
  const __bf16 *Whi, *Wlo;
  __bf16 *Ohi, *Olo;
  if (z == 0) { Whi = (const __bf16*)(ws + O_WQHI); Wlo = (const __bf16*)(ws + O_WQLO);
                Ohi = (__bf16*)(ws + O_QHI);  Olo = (__bf16*)(ws + O_QLO); }
  else if (z == 1) { Whi = (const __bf16*)(ws + O_WKHI); Wlo = (const __bf16*)(ws + O_WKLO);
                Ohi = (__bf16*)(ws + O_KHI);  Olo = (__bf16*)(ws + O_KLO); }
  else { Whi = (const __bf16*)(ws + O_WVHI); Wlo = (const __bf16*)(ws + O_WVLO);
                Ohi = (__bf16*)(ws + O_VTHI); Olo = (__bf16*)(ws + O_VTLO); }
  bool tr = (z == 2);

  int lane = threadIdx.x & 31, wave = threadIdx.x >> 5;
  int row0 = blockIdx.x * 128 + (wave >> 1) * 32;
  int col0 = blockIdx.y * 128 + (wave & 1) * 64;

  v8f acc[2][4] = {};
  for (int k0 = 0; k0 < DIMM; k0 += 32) {
    v16bf aH[2], aL[2];
    for (int r = 0; r < 2; r++) {
      aH[r] = fragA(Xhi, row0 + 16 * r, k0, DIMM, lane);
      aL[r] = fragA(Xlo, row0 + 16 * r, k0, DIMM, lane);
    }
    for (int c = 0; c < 4; c++) {
      v16bf bH = fragB(Whi, col0 + 16 * c, k0, DIMM, lane);
      v16bf bL = fragB(Wlo, col0 + 16 * c, k0, DIMM, lane);
      for (int r = 0; r < 2; r++) {
        acc[r][c] = WMMA_BF16(aH[r], bH, acc[r][c]);
        acc[r][c] = WMMA_BF16(aH[r], bL, acc[r][c]);
        acc[r][c] = WMMA_BF16(aL[r], bH, acc[r][c]);
      }
    }
  }
  int hi8 = (lane & 16) >> 1;
  for (int r = 0; r < 2; r++)
    for (int c = 0; c < 4; c++)
      for (int i = 0; i < 8; i++) {
        int n = row0 + 16 * r + i + hi8;
        int o = col0 + 16 * c + (lane & 15);
        int h = o >> 6, d = o & 63;
        __bf16 hv, lv;
        splitf(acc[r][c][i], hv, lv);
        size_t idx = tr ? ((size_t)h * HD + d) * NTOK + n
                        : ((size_t)h * NTOK + n) * HD + d;
        Ohi[idx] = hv;
        Olo[idx] = lv;
      }
}

// ---------------------------------------------------------------------------
// 2) phi(q|k): proj = q . omega^T (WMMA, K=64); phi = exp(proj - 0.5|q|^2)/8.
//    z=0 -> phi_q [h][n][m]; z=1 -> phi_k [h][n][m] + transposed [h][m][n].
// ---------------------------------------------------------------------------
__global__ void __launch_bounds__(128) k_phi(char* ws) {
  int which = blockIdx.z;
  const __bf16* Shi = (const __bf16*)(ws + (which ? O_KHI : O_QHI));
  const __bf16* Slo = (const __bf16*)(ws + (which ? O_KLO : O_QLO));
  const __bf16* OmH = (const __bf16*)(ws + O_OMHI);
  const __bf16* OmL = (const __bf16*)(ws + O_OMLO);
  __bf16* Phi = (__bf16*)(ws + (which ? O_PKHI : O_PQHI));
  __bf16* Plo = (__bf16*)(ws + (which ? O_PKLO : O_PQLO));
  __bf16* PTh = which ? (__bf16*)(ws + O_PKTHI) : nullptr;
  __bf16* PTl = which ? (__bf16*)(ws + O_PKTLO) : nullptr;

  int h = blockIdx.y;
  int lane = threadIdx.x & 31, wave = threadIdx.x >> 5;
  int n0 = blockIdx.x * 128 + wave * 32;
  const __bf16* qh = Shi + (size_t)h * NTOK * HD;
  const __bf16* ql = Slo + (size_t)h * NTOK * HD;

  __shared__ float nrm[128];
  {
    int n = n0 + lane;
    const __bf16* ph = qh + (size_t)n * HD;
    const __bf16* pl = ql + (size_t)n * HD;
    float s = 0.f;
    for (int i = 0; i < HD; i++) {
      float f = (float)ph[i] + (float)pl[i];
      s += f * f;
    }
    nrm[wave * 32 + lane] = 0.5f * s;
  }
  __syncthreads();

  v8f acc[2][4] = {};
  for (int k0 = 0; k0 < HD; k0 += 32) {
    v16bf aH[2], aL[2];
    for (int r = 0; r < 2; r++) {
      aH[r] = fragA(qh, n0 + 16 * r, k0, HD, lane);
      aL[r] = fragA(ql, n0 + 16 * r, k0, HD, lane);
    }
    for (int c = 0; c < 4; c++) {
      v16bf bH = fragB(OmH, 16 * c, k0, HD, lane);
      v16bf bL = fragB(OmL, 16 * c, k0, HD, lane);
      for (int r = 0; r < 2; r++) {
        acc[r][c] = WMMA_BF16(aH[r], bH, acc[r][c]);
        acc[r][c] = WMMA_BF16(aH[r], bL, acc[r][c]);
        acc[r][c] = WMMA_BF16(aL[r], bH, acc[r][c]);
      }
    }
  }
  int hi8 = (lane & 16) >> 1;
  for (int r = 0; r < 2; r++)
    for (int c = 0; c < 4; c++)
      for (int i = 0; i < 8; i++) {
        int lr = 16 * r + i + hi8;
        int n = n0 + lr;
        int m = 16 * c + (lane & 15);
        float val = expf(acc[r][c][i] - nrm[wave * 32 + lr]) * 0.125f; // 1/sqrt(64)
        __bf16 hv, lv;
        splitf(val, hv, lv);
        size_t idx = ((size_t)h * NTOK + n) * NF + m;
        Phi[idx] = hv;
        Plo[idx] = lv;
        if (PTh) {
          size_t t = ((size_t)h * NF + m) * NTOK + n;
          PTh[t] = hv;
          PTl[t] = lv;
        }
      }
}

// ---------------------------------------------------------------------------
// 3) pass1: per (chunk, head) KV_c[m][d] = sum_{n in chunk} phi_k[n][m]*v[n][d]
//    (= phi_k^T . v, both operands available row-major transposed), plus
//    chunk column sums of phi_k.
// ---------------------------------------------------------------------------
__global__ void __launch_bounds__(256) k_chunkkv(char* ws) {
  const __bf16* AH = (const __bf16*)(ws + O_PKTHI);
  const __bf16* AL = (const __bf16*)(ws + O_PKTLO);
  const __bf16* BH = (const __bf16*)(ws + O_VTHI);
  const __bf16* BL = (const __bf16*)(ws + O_VTLO);
  float* CKV = (float*)(ws + O_CKV);
  float* CKS = (float*)(ws + O_CKS);
  int c = blockIdx.x, h = blockIdx.y;
  int lane = threadIdx.x & 31, wave = threadIdx.x >> 5;
  int m0 = (wave >> 2) * 32, d0 = (wave & 3) * 16;
  int k0 = c * CHK;

  const __bf16* ah = AH + (size_t)h * NF * NTOK;
  const __bf16* al = AL + (size_t)h * NF * NTOK;
  v8f acc[2] = {};
  v16bf bH = fragB(BH + (size_t)h * HD * NTOK, d0, k0, NTOK, lane);
  v16bf bL = fragB(BL + (size_t)h * HD * NTOK, d0, k0, NTOK, lane);
  for (int r = 0; r < 2; r++) {
    v16bf aH = fragA(ah, m0 + 16 * r, k0, NTOK, lane);
    v16bf aL2 = fragA(al, m0 + 16 * r, k0, NTOK, lane);
    acc[r] = WMMA_BF16(aH, bH, acc[r]);
    acc[r] = WMMA_BF16(aH, bL, acc[r]);
    acc[r] = WMMA_BF16(aL2, bH, acc[r]);
  }
  int hi8 = (lane & 16) >> 1;
  float* out = CKV + ((size_t)h * NCHK + c) * NF * HD;
  for (int r = 0; r < 2; r++)
    for (int i = 0; i < 8; i++) {
      int m = m0 + 16 * r + i + hi8;
      int d = d0 + (lane & 15);
      out[m * HD + d] = acc[r][i];
    }
  if (threadIdx.x < NF) {
    int m = threadIdx.x;
    const __bf16* rh = ah + (size_t)m * NTOK + k0;
    const __bf16* rl = al + (size_t)m * NTOK + k0;
    float s = 0.f;
    for (int j = 0; j < CHK; j++) s += (float)rh[j] + (float)rl[j];
    CKS[((size_t)h * NCHK + c) * NF + m] = s;
  }
}

// ---------------------------------------------------------------------------
// 4) pass2: per head, exclusive prefix over 64 chunks. Emits KV^T prefix
//    (bf16 hi/lo, layout [h][c][d][m] for WMMA B-side) and ksum prefix (f32).
// ---------------------------------------------------------------------------
__global__ void __launch_bounds__(256) k_prefix(char* ws) {
  const float* CKV = (const float*)(ws + O_CKV);
  const float* CKS = (const float*)(ws + O_CKS);
  __bf16* KH = (__bf16*)(ws + O_KVTHI);
  __bf16* KL = (__bf16*)(ws + O_KVTLO);
  float* KSP = (float*)(ws + O_KSP);
  int h = blockIdx.x, tid = threadIdx.x;
  float st[16];
  for (int j = 0; j < 16; j++) st[j] = 0.f;
  float ss = 0.f;
  for (int c = 0; c < NCHK; c++) {
    size_t base = ((size_t)h * NCHK + c) * NF * HD;
    for (int j = 0; j < 16; j++) {
      int e = tid + 256 * j;
      int m = e >> 6, d = e & 63;
      __bf16 hv, lv;
      splitf(st[j], hv, lv);
      size_t t = base + (size_t)d * NF + m;
      KH[t] = hv;
      KL[t] = lv;
      st[j] += CKV[base + e];
    }
    if (tid < NF) {
      KSP[((size_t)h * NCHK + c) * NF + tid] = ss;
      ss += CKS[((size_t)h * NCHK + c) * NF + tid];
    }
  }
}

// ---------------------------------------------------------------------------
// 5) pass3: per (chunk, head):
//    inter = phi_q . KV_prefix (WMMA), S = phi_q . phi_k^T (WMMA),
//    mask S causally -> LDS -> reload as A-fragments (hi/lo) -> + S.v (WMMA),
//    z = phi_q . ksum_prefix + rowsum(masked S); out = (inter+intra)/ (z+1e-6).
//    Attn written bf16 hi/lo row-major [n][h*64+d].
// ---------------------------------------------------------------------------
__global__ void __launch_bounds__(256) k_pass3(char* ws) {
  const __bf16* PQH = (const __bf16*)(ws + O_PQHI);
  const __bf16* PQL = (const __bf16*)(ws + O_PQLO);
  const __bf16* PKH = (const __bf16*)(ws + O_PKHI);
  const __bf16* PKL = (const __bf16*)(ws + O_PKLO);
  const __bf16* VTH = (const __bf16*)(ws + O_VTHI);
  const __bf16* VTL = (const __bf16*)(ws + O_VTLO);
  const __bf16* KVH = (const __bf16*)(ws + O_KVTHI);
  const __bf16* KVL = (const __bf16*)(ws + O_KVTLO);
  const float* KSP = (const float*)(ws + O_KSP);
  __bf16* ATH = (__bf16*)(ws + O_ATHI);
  __bf16* ATL = (__bf16*)(ws + O_ATLO);

  int lane = threadIdx.x & 31, wave = threadIdx.x >> 5;
  int c = blockIdx.x * 8 + wave, h = blockIdx.y;
  int n0 = c * CHK;

  __shared__ float S[8][32][36]; // 36-float row stride: 16B-aligned rows
  __shared__ float zr[8][32];

  const __bf16* pqh = PQH + (size_t)h * NTOK * NF;
  const __bf16* pql = PQL + (size_t)h * NTOK * NF;
  const __bf16* pkh = PKH + (size_t)h * NTOK * NF;
  const __bf16* pkl = PKL + (size_t)h * NTOK * NF;
  const __bf16* kvh = KVH + ((size_t)h * NCHK + c) * NF * HD;
  const __bf16* kvl = KVL + ((size_t)h * NCHK + c) * NF * HD;
  const __bf16* vth = VTH + (size_t)h * HD * NTOK;
  const __bf16* vtl = VTL + (size_t)h * HD * NTOK;

  v8f acc[2][4] = {};
  v8f sacc[2][2] = {};
  for (int k0 = 0; k0 < NF; k0 += 32) {
    v16bf aH[2], aL[2];
    for (int r = 0; r < 2; r++) {
      aH[r] = fragA(pqh, n0 + 16 * r, k0, NF, lane);
      aL[r] = fragA(pql, n0 + 16 * r, k0, NF, lane);
    }
    for (int cc = 0; cc < 4; cc++) { // inter: B rows = KV^T rows (d), K=m
      v16bf bH = fragB(kvh, 16 * cc, k0, NF, lane);
      v16bf bL = fragB(kvl, 16 * cc, k0, NF, lane);
      for (int r = 0; r < 2; r++) {
        acc[r][cc] = WMMA_BF16(aH[r], bH, acc[r][cc]);
        acc[r][cc] = WMMA_BF16(aH[r], bL, acc[r][cc]);
        acc[r][cc] = WMMA_BF16(aL[r], bH, acc[r][cc]);
      }
    }
    for (int cc = 0; cc < 2; cc++) { // S: B rows = phi_k rows (n'), K=m
      v16bf bH = fragB(pkh, n0 + 16 * cc, k0, NF, lane);
      v16bf bL = fragB(pkl, n0 + 16 * cc, k0, NF, lane);
      for (int r = 0; r < 2; r++) {
        sacc[r][cc] = WMMA_BF16(aH[r], bH, sacc[r][cc]);
        sacc[r][cc] = WMMA_BF16(aH[r], bL, sacc[r][cc]);
        sacc[r][cc] = WMMA_BF16(aL[r], bH, sacc[r][cc]);
      }
    }
  }
  int hi8 = (lane & 16) >> 1;
  for (int r = 0; r < 2; r++)
    for (int cc = 0; cc < 2; cc++)
      for (int i = 0; i < 8; i++) {
        int nl = 16 * r + i + hi8, np = 16 * cc + (lane & 15);
        S[wave][nl][np] = (np <= nl) ? sacc[r][cc][i] : 0.f;
      }
  __syncthreads();
  { // z normalizer: one token row per lane
    int n = n0 + lane;
    const float* ks = KSP + ((size_t)h * NCHK + c) * NF;
    const __bf16* rh = pqh + (size_t)n * NF;
    const __bf16* rl = pql + (size_t)n * NF;
    float zz = 0.f;
    for (int m = 0; m < NF; m++) zz += ((float)rh[m] + (float)rl[m]) * ks[m];
    for (int np = 0; np < CHK; np++) zz += S[wave][lane][np];
    zr[wave][lane] = 1.0f / (zz + 1e-6f);
  }
  __syncthreads();
  { // intra: masked S (from LDS, split to bf16) . v
    v16bf aH[2], aL[2];
    for (int r = 0; r < 2; r++) {
      const float* p = &S[wave][16 * r + (lane & 15)][(lane & 16) >> 1];
      float f[16];
      *(f32x4*)(f)      = *(const f32x4*)(p);
      *(f32x4*)(f + 4)  = *(const f32x4*)(p + 4);
      *(f32x4*)(f + 8)  = *(const f32x4*)(p + 16);
      *(f32x4*)(f + 12) = *(const f32x4*)(p + 20);
      union { __bf16 b[16]; v16bf v; } uh, ul;
      for (int t = 0; t < 16; t++) splitf(f[t], uh.b[t], ul.b[t]);
      aH[r] = uh.v;
      aL[r] = ul.v;
    }
    for (int cc = 0; cc < 4; cc++) { // B rows = v^T rows (d), K = n' (chunk-local)
      v16bf bH = fragB(vth, 16 * cc, n0, NTOK, lane);
      v16bf bL = fragB(vtl, 16 * cc, n0, NTOK, lane);
      for (int r = 0; r < 2; r++) {
        acc[r][cc] = WMMA_BF16(aH[r], bH, acc[r][cc]);
        acc[r][cc] = WMMA_BF16(aH[r], bL, acc[r][cc]);
        acc[r][cc] = WMMA_BF16(aL[r], bH, acc[r][cc]);
      }
    }
  }
  for (int r = 0; r < 2; r++)
    for (int cc = 0; cc < 4; cc++)
      for (int i = 0; i < 8; i++) {
        int nl = 16 * r + i + hi8;
        int n = n0 + nl;
        int d = 16 * cc + (lane & 15);
        float f = acc[r][cc][i] * zr[wave][nl];
        __bf16 hv, lv;
        splitf(f, hv, lv);
        size_t idx = (size_t)n * DIMM + h * HD + d;
        ATH[idx] = hv;
        ATL[idx] = lv;
      }
}

// ---------------------------------------------------------------------------
// 6) Output projection: y = attn . Wo^T + bo, f32 result to d_out.
// ---------------------------------------------------------------------------
__global__ void __launch_bounds__(256) k_gemm_out(char* ws, const float* __restrict__ bias,
                                                 float* __restrict__ out) {
  const __bf16* Ahi = (const __bf16*)(ws + O_ATHI);
  const __bf16* Alo = (const __bf16*)(ws + O_ATLO);
  const __bf16* Whi = (const __bf16*)(ws + O_WOHI);
  const __bf16* Wlo = (const __bf16*)(ws + O_WOLO);
  int lane = threadIdx.x & 31, wave = threadIdx.x >> 5;
  int row0 = blockIdx.x * 128 + (wave >> 1) * 32;
  int col0 = blockIdx.y * 128 + (wave & 1) * 64;
  v8f acc[2][4] = {};
  for (int k0 = 0; k0 < DIMM; k0 += 32) {
    v16bf aH[2], aL[2];
    for (int r = 0; r < 2; r++) {
      aH[r] = fragA(Ahi, row0 + 16 * r, k0, DIMM, lane);
      aL[r] = fragA(Alo, row0 + 16 * r, k0, DIMM, lane);
    }
    for (int cc = 0; cc < 4; cc++) {
      v16bf bH = fragB(Whi, col0 + 16 * cc, k0, DIMM, lane);
      v16bf bL = fragB(Wlo, col0 + 16 * cc, k0, DIMM, lane);
      for (int r = 0; r < 2; r++) {
        acc[r][cc] = WMMA_BF16(aH[r], bH, acc[r][cc]);
        acc[r][cc] = WMMA_BF16(aH[r], bL, acc[r][cc]);
        acc[r][cc] = WMMA_BF16(aL[r], bH, acc[r][cc]);
      }
    }
  }
  int hi8 = (lane & 16) >> 1;
  for (int r = 0; r < 2; r++)
    for (int cc = 0; cc < 4; cc++)
      for (int i = 0; i < 8; i++) {
        int n = row0 + 16 * r + i + hi8;
        int o = col0 + 16 * cc + (lane & 15);
        out[(size_t)n * DIMM + o] = acc[r][cc][i] + bias[o];
      }
}

// ---------------------------------------------------------------------------
extern "C" void kernel_launch(void* const* d_in, const int* in_sizes, int n_in,
                              void* d_out, int out_size, void* d_ws, size_t ws_size,
                              hipStream_t stream) {
  (void)in_sizes; (void)n_in; (void)out_size;
  if (ws_size < WS_NEEDED) return; // need ~73 MB scratch

  const float* x  = (const float*)d_in[0];
  const float* Wq = (const float*)d_in[1];
  const float* Wk = (const float*)d_in[2];
  const float* Wv = (const float*)d_in[3];
  const float* Wo = (const float*)d_in[4];
  const float* bo = (const float*)d_in[5];
  const float* om = (const float*)d_in[6];
  char* ws = (char*)d_ws;

  auto conv = [&](const float* s, size_t off_hi, size_t off_lo, size_t n) {
    int blk = 256;
    int grd = (int)((n + blk - 1) / blk);
    k_convert<<<grd, blk, 0, stream>>>(s, (__bf16*)(ws + off_hi), (__bf16*)(ws + off_lo), (int)n);
  };
  conv(x,  O_XHI,  O_XLO,  E_X);
  conv(Wq, O_WQHI, O_WQLO, E_W);
  conv(Wk, O_WKHI, O_WKLO, E_W);
  conv(Wv, O_WVHI, O_WVLO, E_W);
  conv(Wo, O_WOHI, O_WOLO, E_W);
  conv(om, O_OMHI, O_OMLO, E_OM);

  k_gemm_qkv<<<dim3(16, 8, 3), 256, 0, stream>>>(ws);
  k_phi<<<dim3(16, 16, 2), 128, 0, stream>>>(ws);
  k_chunkkv<<<dim3(NCHK, NH), 256, 0, stream>>>(ws);
  k_prefix<<<NH, 256, 0, stream>>>(ws);
  k_pass3<<<dim3(NCHK / 8, NH), 256, 0, stream>>>(ws);
  k_gemm_out<<<dim3(16, 8), 256, 0, stream>>>(ws, bo, (float*)d_out);
}
